// GNN_70978629533708
// MI455X (gfx1250) — compile-verified, matrix operand order checked by reading, not run on hardware
//
#include <hip/hip_runtime.h>

#define D_DIM 256

typedef __attribute__((ext_vector_type(16))) __bf16 v16bf;
typedef __attribute__((ext_vector_type(8)))  __bf16 v8bf;
typedef __attribute__((ext_vector_type(8)))  float  v8f;

__device__ __forceinline__ unsigned short f2bf(float f) {
    unsigned u = __float_as_uint(f);
    unsigned r = u + 0x7FFFu + ((u >> 16) & 1u);   // round-to-nearest-even
    return (unsigned short)(r >> 16);
}

// ---------------- degree histogram (int) ----------------
__global__ void k_count(const int* __restrict__ dst, int* __restrict__ cnt, int E) {
    int i = blockIdx.x * blockDim.x + threadIdx.x;
    if (i < E) atomicAdd(&cnt[dst[i]], 1);
}

__global__ void k_dinv(const int* __restrict__ cnt, float* __restrict__ dinv, int N) {
    int i = blockIdx.x * blockDim.x + threadIdx.x;
    if (i < N) dinv[i] = rsqrtf((float)cnt[i] + 1.0f);   // + self loop
}

// ---------------- CSR build: block-wise scan ----------------
__global__ void k_scan1(const int* __restrict__ cnt, int* __restrict__ incl,
                        int* __restrict__ bsum, int N) {
    __shared__ int sh[256];
    int i = blockIdx.x * 256 + threadIdx.x;
    int v = (i < N) ? cnt[i] : 0;
    sh[threadIdx.x] = v;
    __syncthreads();
#pragma unroll
    for (int off = 1; off < 256; off <<= 1) {
        int t = (threadIdx.x >= off) ? sh[threadIdx.x - off] : 0;
        __syncthreads();
        sh[threadIdx.x] += t;
        __syncthreads();
    }
    if (i < N) incl[i] = sh[threadIdx.x];
    if (threadIdx.x == 255) bsum[blockIdx.x] = sh[255];
}

__global__ void k_scan2(int* __restrict__ bsum, int nb) {   // serial exclusive scan (tiny)
    if (blockIdx.x == 0 && threadIdx.x == 0) {
        int run = 0;
        for (int i = 0; i < nb; ++i) { int v = bsum[i]; bsum[i] = run; run += v; }
    }
}

__global__ void k_scan3(const int* __restrict__ incl, const int* __restrict__ cnt,
                        const int* __restrict__ boff, int* __restrict__ row_ptr,
                        int N, int E) {
    int i = blockIdx.x * blockDim.x + threadIdx.x;
    if (i < N) row_ptr[i] = boff[i >> 8] + incl[i] - cnt[i];   // exclusive
    if (i == N) row_ptr[N] = E;
}

__global__ void k_fill(const int* __restrict__ src, const int* __restrict__ dst,
                       const int* __restrict__ row_ptr, int* __restrict__ cursor,
                       const float* __restrict__ dinv, int* __restrict__ srcs,
                       float* __restrict__ coef, int E) {
    int e = blockIdx.x * blockDim.x + threadIdx.x;
    if (e >= E) return;
    int s = src[e], d = dst[e];
    int slot = row_ptr[d] + atomicAdd(&cursor[d], 1);
    srcs[slot] = s;
    coef[slot] = dinv[s] * dinv[d];
}

// ---------------- weight transpose + bf16 convert ----------------
__global__ void k_wt(const float* __restrict__ W, unsigned short* __restrict__ Wt) {
    int idx = blockIdx.x * blockDim.x + threadIdx.x;
    if (idx < D_DIM * D_DIM) {
        int n = idx >> 8, k = idx & 255;
        Wt[idx] = f2bf(W[k * D_DIM + n]);
    }
}

// activations f32 -> bf16, optional fused ReLU
__global__ void k_cvt(const float* __restrict__ H, unsigned short* __restrict__ A,
                      long total, int relu) {
    long i = ((long)blockIdx.x * blockDim.x + threadIdx.x) * 4;
    if (i >= total) return;
    float4 v = *(const float4*)(H + i);
    if (relu) {
        v.x = fmaxf(v.x, 0.0f); v.y = fmaxf(v.y, 0.0f);
        v.z = fmaxf(v.z, 0.0f); v.w = fmaxf(v.w, 0.0f);
    }
    ushort4 o;
    o.x = f2bf(v.x); o.y = f2bf(v.y); o.z = f2bf(v.z); o.w = f2bf(v.w);
    *(ushort4*)(A + i) = o;
}

// ---------------- WMMA GEMM with LDS-staged B tile ----------------
// Block: 256 threads = 8 waves. Wave w -> rows [blk.x*128 + w*16, +16),
// cols [blk.y*64, +64). B tile (64 x 256 bf16 = 32KB) staged in LDS.
__global__ __launch_bounds__(256) void k_gemm(const unsigned short* __restrict__ A,
                                              const unsigned short* __restrict__ Wt,
                                              float* __restrict__ C, int M) {
    __shared__ unsigned short sB[64 * D_DIM];   // 32 KB

    const int tid  = threadIdx.x;
    const int lane = tid & 31;
    const int wave = tid >> 5;
    const int base_m = blockIdx.x * 128 + wave * 16;
    const int base_n = blockIdx.y * 64;

    // cooperative stage of Wt[base_n .. base_n+63][:] -> 8x uint4 per thread
    const unsigned short* wsrc = Wt + (size_t)base_n * D_DIM;
#pragma unroll
    for (int j = 0; j < 8; ++j) {
        int idx = (j * 256 + tid) * 8;          // in shorts, 16B granules
        *(uint4*)(sB + idx) = *(const uint4*)(wsrc + idx);
    }
    __syncthreads();

    if (base_m >= M) return;

    const int half = lane >> 4;      // 0: K{0-7,16-23}; 1: K{8-15,24-31}
    const int r    = lane & 15;

    const unsigned short* pa = A + (size_t)(base_m + r) * D_DIM + half * 8;

    v8f acc[4] = {v8f{}, v8f{}, v8f{}, v8f{}};

#pragma unroll
    for (int kb = 0; kb < D_DIM; kb += 32) {
        v8bf a0 = *(const v8bf*)(pa + kb);
        v8bf a1 = *(const v8bf*)(pa + kb + 16);
        v16bf a;
#pragma unroll
        for (int j = 0; j < 8; ++j) { a[j] = a0[j]; a[j + 8] = a1[j]; }

#pragma unroll
        for (int t = 0; t < 4; ++t) {
            const unsigned short* pb = sB + (t * 16 + r) * D_DIM + kb + half * 8;
            v8bf b0 = *(const v8bf*)(pb);
            v8bf b1 = *(const v8bf*)(pb + 16);
            v16bf b;
#pragma unroll
            for (int j = 0; j < 8; ++j) { b[j] = b0[j]; b[j + 8] = b1[j]; }
            acc[t] = __builtin_amdgcn_wmma_f32_16x16x32_bf16(
                false, a, false, b, (short)0, acc[t], false, false);
        }
    }

    // C/D layout: VGPR v, lanes 0-15 -> row v, lanes 16-31 -> row v+8; col = lane&15
#pragma unroll
    for (int t = 0; t < 4; ++t) {
        int col = base_n + t * 16 + r;
        float* pc = C + (size_t)(base_m + half * 8) * D_DIM + col;
#pragma unroll
        for (int v = 0; v < 8; ++v) pc[(size_t)v * D_DIM] = acc[t][v];
    }
}

// ---------------- fused aggregation: gather + self-loop + bias ----------------
// One wave per destination node; lane owns 8 channels. No float atomics.
__global__ __launch_bounds__(256) void k_agg(const int* __restrict__ row_ptr,
                                             const int* __restrict__ srcs,
                                             const float* __restrict__ coef,
                                             const float* __restrict__ Hlin,
                                             const float* __restrict__ dinv,
                                             const float* __restrict__ bias,
                                             float* __restrict__ Hout, int N) {
    int node = (int)((blockIdx.x * (long)blockDim.x + threadIdx.x) >> 5);
    int lane = threadIdx.x & 31;
    if (node >= N) return;
    const int c0 = lane * 8;

    float dv = dinv[node];
    float s2 = dv * dv;
    const float* ph = Hlin + (size_t)node * D_DIM + c0;
    float4 h0 = *(const float4*)(ph);
    float4 h1 = *(const float4*)(ph + 4);
    float4 b0 = *(const float4*)(bias + c0);
    float4 b1 = *(const float4*)(bias + c0 + 4);

    float4 a0, a1;
    a0.x = s2 * h0.x + b0.x; a0.y = s2 * h0.y + b0.y;
    a0.z = s2 * h0.z + b0.z; a0.w = s2 * h0.w + b0.w;
    a1.x = s2 * h1.x + b1.x; a1.y = s2 * h1.y + b1.y;
    a1.z = s2 * h1.z + b1.z; a1.w = s2 * h1.w + b1.w;

    int beg = row_ptr[node], end = row_ptr[node + 1];
    for (int e = beg; e < end; ++e) {
        int   si = srcs[e];
        float cf = coef[e];
        const float* ps = Hlin + (size_t)si * D_DIM + c0;
        float4 m0 = *(const float4*)(ps);
        float4 m1 = *(const float4*)(ps + 4);
        a0.x = fmaf(cf, m0.x, a0.x); a0.y = fmaf(cf, m0.y, a0.y);
        a0.z = fmaf(cf, m0.z, a0.z); a0.w = fmaf(cf, m0.w, a0.w);
        a1.x = fmaf(cf, m1.x, a1.x); a1.y = fmaf(cf, m1.y, a1.y);
        a1.z = fmaf(cf, m1.z, a1.z); a1.w = fmaf(cf, m1.w, a1.w);
    }

    float* po = Hout + (size_t)node * D_DIM + c0;
    *(float4*)(po)     = a0;
    *(float4*)(po + 4) = a1;
}

// ---------------- mean pooling ----------------
__global__ void k_cnt(const int* __restrict__ batch, float* __restrict__ cnt, int N) {
    int i = blockIdx.x * blockDim.x + threadIdx.x;
    if (i < N) atomicAdd(&cnt[batch[i]], 1.0f);
}

__global__ void k_pool(const float* __restrict__ z, const int* __restrict__ batch,
                       float* __restrict__ out, int N) {
    long idx = (long)blockIdx.x * blockDim.x + threadIdx.x;
    if (idx >= (long)N * 64) return;
    int node = (int)(idx >> 6);
    int c4 = (int)(idx & 63) * 4;
    int g = batch[node];
    float4 v = *(const float4*)(z + (size_t)node * D_DIM + c4);
    float* po = out + (size_t)g * D_DIM + c4;
    atomicAdd(po + 0, v.x); atomicAdd(po + 1, v.y);
    atomicAdd(po + 2, v.z); atomicAdd(po + 3, v.w);
}

__global__ void k_div(float* __restrict__ out, const float* __restrict__ cnt, int total) {
    int idx = blockIdx.x * blockDim.x + threadIdx.x;
    if (idx < total) out[idx] /= fmaxf(cnt[idx >> 8], 1.0f);
}

// ---------------- launcher ----------------
static inline size_t align256(size_t x) { return (x + 255) & ~(size_t)255; }
static inline int cdiv(long a, long b) { return (int)((a + b - 1) / b); }

extern "C" void kernel_launch(void* const* d_in, const int* in_sizes, int n_in,
                              void* d_out, int out_size, void* d_ws, size_t ws_size,
                              hipStream_t stream) {
    const float* x     = (const float*)d_in[0];
    const int*   ei    = (const int*)d_in[1];
    const int*   batch = (const int*)d_in[2];
    const float* W[3]  = {(const float*)d_in[4], (const float*)d_in[6], (const float*)d_in[8]};
    const float* B[3]  = {(const float*)d_in[5], (const float*)d_in[7], (const float*)d_in[9]};
    float* out = (float*)d_out;

    const int N = in_sizes[0] / D_DIM;
    const int E = in_sizes[1] / 2;
    const int G = out_size / D_DIM;
    const int* srcp = ei;
    const int* dstp = ei + E;
    const long nd = (long)N * D_DIM;
    const int nb = cdiv(N, 256);

    // workspace carve-out
    char* p = (char*)d_ws;
    float* dinv   = (float*)p; p += align256((size_t)N * 4);
    float* gcnt   = (float*)p; p += align256((size_t)G * 4);
    int*   cnti   = (int*)p;   p += align256((size_t)N * 4);
    int*   incl   = (int*)p;   p += align256((size_t)N * 4);
    int*   bsum   = (int*)p;   p += align256((size_t)nb * 4);
    int*   rowp   = (int*)p;   p += align256((size_t)(N + 1) * 4);
    int*   cursor = (int*)p;   p += align256((size_t)N * 4);
    int*   srcs   = (int*)p;   p += align256((size_t)E * 4);
    float* coef   = (float*)p; p += align256((size_t)E * 4);
    float* Hlin   = (float*)p; p += align256((size_t)nd * 4);
    float* H0     = (float*)p; p += align256((size_t)nd * 4);
    float* H1     = (float*)p; p += align256((size_t)nd * 4);
    unsigned short* Abf = (unsigned short*)p; p += align256((size_t)nd * 2);
    unsigned short* Wt[3];
    for (int l = 0; l < 3; ++l) { Wt[l] = (unsigned short*)p; p += align256((size_t)D_DIM * D_DIM * 2); }

    const int TB = 256;

    // degree + normalization + CSR build
    hipMemsetAsync(cnti, 0, (size_t)N * 4, stream);
    hipMemsetAsync(cursor, 0, (size_t)N * 4, stream);
    k_count<<<cdiv(E, TB), TB, 0, stream>>>(dstp, cnti, E);
    k_dinv<<<cdiv(N, TB), TB, 0, stream>>>(cnti, dinv, N);
    k_scan1<<<nb, TB, 0, stream>>>(cnti, incl, bsum, N);
    k_scan2<<<1, 1, 0, stream>>>(bsum, nb);
    k_scan3<<<cdiv(N + 1, TB), TB, 0, stream>>>(incl, cnti, bsum, rowp, N, E);
    k_fill<<<cdiv(E, TB), TB, 0, stream>>>(srcp, dstp, rowp, cursor, dinv, srcs, coef, E);

    // weights -> bf16 transposed
    for (int l = 0; l < 3; ++l)
        k_wt<<<cdiv(D_DIM * D_DIM, TB), TB, 0, stream>>>(W[l], Wt[l]);

    dim3 ggrid(cdiv(N, 128), D_DIM / 64);
    const int cvtBlocks = cdiv(nd / 4, TB);
    const int aggBlocks = cdiv((long)N * 32, TB);

    const float* lin_in[3] = {x, H0, H1};
    float*       lout[3]   = {H0, H1, H0};

    for (int l = 0; l < 3; ++l) {
        k_cvt<<<cvtBlocks, TB, 0, stream>>>(lin_in[l], Abf, nd, l > 0 ? 1 : 0);
        k_gemm<<<ggrid, TB, 0, stream>>>(Abf, Wt[l], Hlin, N);
        k_agg<<<aggBlocks, TB, 0, stream>>>(rowp, srcs, coef, Hlin, dinv, B[l], lout[l], N);
    }

    // mean pooling of z = H0
    hipMemsetAsync(gcnt, 0, (size_t)G * 4, stream);
    hipMemsetAsync(out, 0, (size_t)out_size * 4, stream);
    k_cnt<<<cdiv(N, TB), TB, 0, stream>>>(batch, gcnt, N);
    k_pool<<<cdiv((long)N * 64, TB), TB, 0, stream>>>(H0, batch, out, N);
    k_div<<<cdiv(G * D_DIM, TB), TB, 0, stream>>>(out, gcnt, G * D_DIM);
}
